// KANLayer_11776800325919
// MI455X (gfx1250) — matrix operand order repaired and checked by reading, not compiled, not run on hardware
//
#include <hip/hip_runtime.h>
#include <hip/hip_bf16.h>
#include <math.h>

typedef _Float16 v16h __attribute__((ext_vector_type(16)));
typedef _Float16 v8h  __attribute__((ext_vector_type(8)));
typedef float    v8f  __attribute__((ext_vector_type(8)));
typedef unsigned tdm_u4 __attribute__((ext_vector_type(4)));
typedef int      tdm_i8 __attribute__((ext_vector_type(8)));
typedef int      tdm_i4 __attribute__((ext_vector_type(4)));

#define B_DIM 8192
#define I_DIM 256
#define O_DIM 256
#define M_BAS 7
#define K_DIM (I_DIM * 8)   // 2048: [silu, basis0..6] per input channel

// GEMM blocking
#define BM 128              // block M tile
#define BN 64               // block N tile
#define KC 64               // K chunk staged in LDS (halves)
#define LDS_CHUNK (BM * KC + BN * KC)   // halves per buffer (12288 = 24 KB)

#if __has_builtin(__builtin_amdgcn_tensor_load_to_lds)
#define HAVE_TDM 1
#else
#define HAVE_TDM 0
#endif

// ---------------------------------------------------------------------------
// Kernel 1: per-column min/max -> centre, inv_halfwidth
// ---------------------------------------------------------------------------
__global__ __launch_bounds__(256) void kan_colstats(const float* __restrict__ x,
                                                    float* __restrict__ centre,
                                                    float* __restrict__ invhw,
                                                    int Brows) {
    const int col = blockIdx.x;
    const int t   = threadIdx.x;
    float mn =  INFINITY, mx = -INFINITY;
    for (int r = t; r < Brows; r += 256) {
        float v = x[(size_t)r * I_DIM + col];
        mn = fminf(mn, v);
        mx = fmaxf(mx, v);
    }
    __shared__ float smn[256], smx[256];
    smn[t] = mn; smx[t] = mx;
    __syncthreads();
    for (int s = 128; s > 0; s >>= 1) {
        if (t < s) {
            smn[t] = fminf(smn[t], smn[t + s]);
            smx[t] = fmaxf(smx[t], smx[t + s]);
        }
        __syncthreads();
    }
    if (t == 0) {
        float w = fmaxf(smx[0] - smn[0], 0.01f);
        centre[col] = 0.5f * (smx[0] + smn[0]);
        invhw[col]  = 2.0f / w;          // 1 / (0.5*width)
    }
}

// ---------------------------------------------------------------------------
// Kernel 2: build A (B x 2048, f16): per (b,i) pack [silu, basis0..6]
// Knots: t_j = -1 + 0.2*j, j=0..10 ; degree-3 de Boor, NUM_BASIS=7
// ---------------------------------------------------------------------------
__global__ __launch_bounds__(256) void kan_build_a(const float* __restrict__ x,
                                                   const float* __restrict__ centre,
                                                   const float* __restrict__ invhw,
                                                   _Float16* __restrict__ Af) {
    const int idx = blockIdx.x * 256 + threadIdx.x;   // b*I_DIM + i
    const int i   = idx & (I_DIM - 1);
    float xn = (x[idx] - centre[i]) * invhw[i];
    xn = fminf(fmaxf(xn, -2.0f), 2.0f);

    float sig  = 1.0f / (1.0f + __expf(-xn));
    float silu = xn * sig;

    float b[8];
    #pragma unroll
    for (int j = 0; j < 7; ++j) {
        float tl = -1.0f + 0.2f * (float)j;
        float tr = -1.0f + 0.2f * (float)(j + 1);
        b[j] = (xn >= tl && xn < tr) ? 1.0f : 0.0f;
    }
    b[7] = 0.0f;
    // reference edge case: x == KNOTS[NUM_BASIS] (= 0.4) forces last order-0 basis
    if (xn == (-1.0f + 0.2f * 7.0f)) b[6] = 1.0f;

    #pragma unroll
    for (int k = 1; k <= 3; ++k) {
        float invd = 1.0f / (0.2f * (float)k);   // d1 = d2 = k*0.2 > 0 always
        #pragma unroll
        for (int j = 0; j < 7; ++j) {
            float tj   = -1.0f + 0.2f * (float)j;
            float tjk1 = -1.0f + 0.2f * (float)(j + k + 1);
            b[j] = (xn - tj) * invd * b[j] + (tjk1 - xn) * invd * b[j + 1];
        }
    }

    v8h pack;
    pack[0] = (_Float16)silu;
    #pragma unroll
    for (int m = 0; m < 7; ++m) pack[1 + m] = (_Float16)b[m];
    *(v8h*)(Af + (size_t)idx * 8) = pack;
}

// ---------------------------------------------------------------------------
// Kernel 3: build W (O x 2048, f16): per (o,i) pack [base_w, coeff*spline_w]
// ---------------------------------------------------------------------------
__global__ __launch_bounds__(256) void kan_build_w(const float* __restrict__ base_w,
                                                   const float* __restrict__ spline_w,
                                                   const float* __restrict__ spline_c,
                                                   _Float16* __restrict__ Wf) {
    const int idx = blockIdx.x * 256 + threadIdx.x;   // o*I_DIM + i
    float sw = spline_w[idx];
    const float* c = spline_c + (size_t)idx * M_BAS;
    v8h pack;
    pack[0] = (_Float16)base_w[idx];
    #pragma unroll
    for (int m = 0; m < M_BAS; ++m) pack[1 + m] = (_Float16)(c[m] * sw);
    *(v8h*)(Wf + (size_t)idx * 8) = pack;
}

// ---------------------------------------------------------------------------
// Copy-engine helpers
// ---------------------------------------------------------------------------
#if HAVE_TDM
// Issue one TDM 2D-tile load: tile = KC x tileRows (2-byte elems), row stride
// K_DIM elems in memory, packed contiguously (row-major, KC pitch) in LDS.
// D# per ISA 8.3/8.4: group0 {count=1, lds_addr, global_addr, type=2},
// group1 {data_size=1(2B), huge tensor dims (tiles never clip), tile dims,
// tensor_dim0_stride}. Groups 2/3 zero (2D tile; tile_dim2/3 unused).
__device__ __forceinline__ void tdm_issue_2d(unsigned ldsAddr, const _Float16* g,
                                             int tileRows) {
    unsigned long long ga = (unsigned long long)(size_t)g;
    tdm_u4 g0;
    g0[0] = 1u;                                            // count=1, user mode
    g0[1] = ldsAddr;                                       // lds_addr
    g0[2] = (unsigned)ga;                                  // global_addr[31:0]
    g0[3] = (unsigned)((ga >> 32) & 0x01FFFFFFu) | (2u << 30); // addr[56:32]|type=2
    const unsigned td = 1u << 20;                          // tensor_dim0/1: huge
    tdm_i8 g1;
    g1[0] = (int)(1u << 16);                               // data_size=1 -> 2 bytes
    g1[1] = (int)((td & 0xFFFFu) << 16);                   // tensor_dim0[15:0]
    g1[2] = (int)(((td >> 16) & 0xFFFFu) | ((td & 0xFFFFu) << 16));
    g1[3] = (int)(((td >> 16) & 0xFFFFu) | ((unsigned)KC << 16)); // tile_dim0=KC
    g1[4] = (int)(unsigned)tileRows;                       // tile_dim1; tile_dim2=0
    g1[5] = (int)K_DIM;                                    // tensor_dim0_stride
    g1[6] = 0;
    g1[7] = 0;
    const tdm_i4 z4 = {0, 0, 0, 0};
#if __clang_major__ >= 23
    const tdm_i8 z8 = {0, 0, 0, 0, 0, 0, 0, 0};
    __builtin_amdgcn_tensor_load_to_lds(g0, g1, z4, z4, z8, 0);
#else
    __builtin_amdgcn_tensor_load_to_lds(g0, g1, z4, z4, 0);
#endif
}

__device__ __forceinline__ void wait_tensor0() {
#if __has_builtin(__builtin_amdgcn_s_wait_tensorcnt)
    __builtin_amdgcn_s_wait_tensorcnt(0);
#else
    asm volatile("s_wait_tensorcnt 0" ::: "memory");
#endif
}
#else  // !HAVE_TDM: per-thread async global->LDS 16B pieces (ASYNCcnt path)
__device__ __forceinline__ void async_copy_b128(unsigned ldsOff, const _Float16* g) {
    asm volatile("global_load_async_to_lds_b128 %0, %1, off"
                 :: "v"(ldsOff), "v"(g) : "memory");
}
__device__ __forceinline__ void wait_async0() {
#if __has_builtin(__builtin_amdgcn_s_wait_asynccnt)
    __builtin_amdgcn_s_wait_asynccnt(0);
#else
    asm volatile("s_wait_asynccnt 0" ::: "memory");
#endif
}
#endif

// ---------------------------------------------------------------------------
// Kernel 4: fused GEMM  out = A(B x K) * W(O x K)^T + bias  via WMMA f16->f32
// Block = 8 waves -> 128(M) x 64(N) tile; waves 2(M) x 4(N), wave = 64x16.
// K streamed in KC=64 chunks, double-buffered in LDS. Chunk fill: one
// tensor_load_to_lds per matrix per chunk (TDM, TENSORcnt), issued by wave 0,
// overlapped with WMMA on the other buffer; workgroup barrier publishes it.
// Fragment layouts as loaded here (wave32; A and B share this K order):
//   A 16x32 f16 : lane h=lane>>4, row=lane&15; 16 contiguous K at K[h*16..+15]
//   B 32x16 f16 : lane col=lane&15; 16 contiguous K at K[h*16..+15] (from W row)
//   C/D 16x16 f32: lane col=lane&15; VGPR v -> row v + 8*h
// ---------------------------------------------------------------------------
__global__ __launch_bounds__(256) void kan_gemm(const _Float16* __restrict__ A,
                                                const _Float16* __restrict__ W,
                                                const float* __restrict__ bias,
                                                float* __restrict__ out) {
    __shared__ _Float16 lds[2][LDS_CHUNK];           // 2 x 24 KB

    const int tid  = threadIdx.x;
    const int lane = tid & 31;
    const int wid  = tid >> 5;
    const int mg   = wid >> 2;           // 0..1  (64-row group)
    const int ng   = wid & 3;            // 0..3  (16-col group)
    const int h    = lane >> 4;
    const int r    = lane & 15;

    const int bn = blockIdx.x & (O_DIM / BN - 1);    // 0..3
    const int bm = blockIdx.x >> 2;                  // M block

    const size_t aRow0 = (size_t)bm * BM;
    const size_t bRow0 = (size_t)bn * BN;

    const unsigned ldsBase0  = (unsigned)(size_t)(&lds[0][0]);
    const unsigned bufStride = (unsigned)(LDS_CHUNK * 2);   // bytes

#if HAVE_TDM
    // Block-uniform tile-start pointers, advanced by KC halves per chunk.
    const _Float16* gAt = A + aRow0 * (size_t)K_DIM;
    const _Float16* gBt = W + bRow0 * (size_t)K_DIM;

    #define ISSUE_CHUNK(BUFOFF)                                          \
        do {                                                             \
            if (wid == 0) {                                              \
                tdm_issue_2d(ldsBase0 + (BUFOFF), gAt, BM);              \
                tdm_issue_2d(ldsBase0 + (BUFOFF) + (unsigned)(BM*KC*2),  \
                             gBt, BN);                                   \
            }                                                            \
            gAt += KC; gBt += KC;                                        \
        } while (0)
    #define WAIT_COPIES()                                                \
        do { if (wid == 0) wait_tensor0(); } while (0)
#else
    // Per-thread cooperative async copy: 6 x 16B pieces per thread per chunk.
    const _Float16* gA0; const _Float16* gA1; const _Float16* gA2; const _Float16* gA3;
    const _Float16* gB0; const _Float16* gB1;
    unsigned lA0, lA1, lA2, lA3, lB0, lB1;
    {
        int p, row, off;
        p = tid;           row = p >> 3; off = (p & 7) * 8;
        gA0 = A + (aRow0 + row) * (size_t)K_DIM + off;
        lA0 = ldsBase0 + (unsigned)((row * KC + off) * 2);
        p = tid + 256;     row = p >> 3; off = (p & 7) * 8;
        gA1 = A + (aRow0 + row) * (size_t)K_DIM + off;
        lA1 = ldsBase0 + (unsigned)((row * KC + off) * 2);
        p = tid + 512;     row = p >> 3; off = (p & 7) * 8;
        gA2 = A + (aRow0 + row) * (size_t)K_DIM + off;
        lA2 = ldsBase0 + (unsigned)((row * KC + off) * 2);
        p = tid + 768;     row = p >> 3; off = (p & 7) * 8;
        gA3 = A + (aRow0 + row) * (size_t)K_DIM + off;
        lA3 = ldsBase0 + (unsigned)((row * KC + off) * 2);
        p = tid;           row = p >> 3; off = (p & 7) * 8;
        gB0 = W + (bRow0 + row) * (size_t)K_DIM + off;
        lB0 = ldsBase0 + (unsigned)((BM * KC + row * KC + off) * 2);
        p = tid + 256;     row = p >> 3; off = (p & 7) * 8;
        gB1 = W + (bRow0 + row) * (size_t)K_DIM + off;
        lB1 = ldsBase0 + (unsigned)((BM * KC + row * KC + off) * 2);
    }
    #define ISSUE_CHUNK(BUFOFF)                                        \
        do {                                                           \
            async_copy_b128(lA0 + (BUFOFF), gA0);                      \
            async_copy_b128(lA1 + (BUFOFF), gA1);                      \
            async_copy_b128(lA2 + (BUFOFF), gA2);                      \
            async_copy_b128(lA3 + (BUFOFF), gA3);                      \
            async_copy_b128(lB0 + (BUFOFF), gB0);                      \
            async_copy_b128(lB1 + (BUFOFF), gB1);                      \
            gA0 += KC; gA1 += KC; gA2 += KC; gA3 += KC;                \
            gB0 += KC; gB1 += KC;                                      \
        } while (0)
    #define WAIT_COPIES() wait_async0()
#endif

    v8f acc0 = {}, acc1 = {}, acc2 = {}, acc3 = {};

    const _Float16* const aFragBase = &lds[0][0] + (mg * 64 + r) * KC + h * 16;
    const _Float16* const bFragBase = &lds[0][BM * KC] + (ng * 16 + r) * KC + h * 16;

    // compute both 32-K steps of the chunk at LDS half-offset BUFHALF
    #define COMPUTE_CHUNK(BUFHALF)                                                   \
        do {                                                                         \
            _Pragma("unroll")                                                        \
            for (int k0 = 0; k0 < KC; k0 += 32) {                                    \
                v16h bf = *(const v16h*)(bFragBase + (BUFHALF) + k0);                \
                const _Float16* ap0 = aFragBase + (BUFHALF) + k0;                    \
                v8h lo, hi; v16h af;                                                 \
                lo = *(const v8h*)(ap0);                                             \
                hi = *(const v8h*)(ap0 + 8);                                         \
                af = __builtin_shufflevector(lo, hi,                                 \
                         0,1,2,3,4,5,6,7,8,9,10,11,12,13,14,15);                     \
                acc0 = __builtin_amdgcn_wmma_f32_16x16x32_f16(                       \
                           false, af, false, bf, (short)0, acc0, false, false);      \
                lo = *(const v8h*)(ap0 + 16 * KC);                                   \
                hi = *(const v8h*)(ap0 + 16 * KC + 8);                               \
                af = __builtin_shufflevector(lo, hi,                                 \
                         0,1,2,3,4,5,6,7,8,9,10,11,12,13,14,15);                     \
                acc1 = __builtin_amdgcn_wmma_f32_16x16x32_f16(                       \
                           false, af, false, bf, (short)0, acc1, false, false);      \
                lo = *(const v8h*)(ap0 + 32 * KC);                                   \
                hi = *(const v8h*)(ap0 + 32 * KC + 8);                               \
                af = __builtin_shufflevector(lo, hi,                                 \
                         0,1,2,3,4,5,6,7,8,9,10,11,12,13,14,15);                     \
                acc2 = __builtin_amdgcn_wmma_f32_16x16x32_f16(                       \
                           false, af, false, bf, (short)0, acc2, false, false);      \
                lo = *(const v8h*)(ap0 + 48 * KC);                                   \
                hi = *(const v8h*)(ap0 + 48 * KC + 8);                               \
                af = __builtin_shufflevector(lo, hi,                                 \
                         0,1,2,3,4,5,6,7,8,9,10,11,12,13,14,15);                     \
                acc3 = __builtin_amdgcn_wmma_f32_16x16x32_f16(                       \
                           false, af, false, bf, (short)0, acc3, false, false);      \
            }                                                                        \
        } while (0)

    // -------- pipeline: prologue fills buffer 0 ----------------------------
    ISSUE_CHUNK(0u);
    WAIT_COPIES();
    __syncthreads();

    const int nchunks = K_DIM / KC;                  // 32 (even)
    for (int c = 0; c < nchunks; c += 2) {
        if (c + 1 < nchunks) ISSUE_CHUNK(bufStride); // chunk c+1 -> buf 1
        COMPUTE_CHUNK(0);                            // chunk c   (buf 0)
        WAIT_COPIES();
        __syncthreads();
        if (c + 2 < nchunks) ISSUE_CHUNK(0u);        // chunk c+2 -> buf 0
        COMPUTE_CHUNK(LDS_CHUNK);                    // chunk c+1 (buf 1)
        WAIT_COPIES();
        __syncthreads();
    }

    #undef ISSUE_CHUNK
    #undef WAIT_COPIES
    #undef COMPUTE_CHUNK

    // -------- store D + bias (lane col = r, VGPR v -> row v + 8*h) ---------
    const int ncol = bn * BN + ng * 16 + r;
    const float bv = bias[ncol];
    float* o0 = out + (aRow0 + (size_t)mg * 64 + 8 * h) * O_DIM + ncol;
    #pragma unroll
    for (int v = 0; v < 8; ++v) o0[(size_t)(v     ) * O_DIM] = acc0[v] + bv;
    #pragma unroll
    for (int v = 0; v < 8; ++v) o0[(size_t)(v + 16) * O_DIM] = acc1[v] + bv;
    #pragma unroll
    for (int v = 0; v < 8; ++v) o0[(size_t)(v + 32) * O_DIM] = acc2[v] + bv;
    #pragma unroll
    for (int v = 0; v < 8; ++v) o0[(size_t)(v + 48) * O_DIM] = acc3[v] + bv;
}

// ---------------------------------------------------------------------------
extern "C" void kernel_launch(void* const* d_in, const int* in_sizes, int n_in,
                              void* d_out, int out_size, void* d_ws, size_t ws_size,
                              hipStream_t stream) {
    const float* x        = (const float*)d_in[0];   // (B, I)
    const float* base_w   = (const float*)d_in[1];   // (O, I)
    const float* spline_w = (const float*)d_in[2];   // (O, I)
    const float* spline_c = (const float*)d_in[3];   // (O, I, 7)
    const float* bias     = (const float*)d_in[4];   // (O,)
    float*       out      = (float*)d_out;           // (B, O)

    const int Brows = in_sizes[0] / I_DIM;           // 8192

    char* ws = (char*)d_ws;
    float*    centre = (float*)ws;                               // 256 f32
    float*    invhw  = centre + I_DIM;                           // 256 f32
    _Float16* Af     = (_Float16*)(ws + 4096);                   // B x 2048 f16
    _Float16* Wf     = (_Float16*)(ws + 4096 +
                           (size_t)Brows * K_DIM * sizeof(_Float16));

    kan_colstats<<<I_DIM, 256, 0, stream>>>(x, centre, invhw, Brows);
    kan_build_a<<<(Brows * I_DIM) / 256, 256, 0, stream>>>(x, centre, invhw, Af);
    kan_build_w<<<(O_DIM * I_DIM) / 256, 256, 0, stream>>>(base_w, spline_w, spline_c, Wf);

    const int nblocks = (Brows / BM) * (O_DIM / BN); // 256
    kan_gemm<<<nblocks, 256, 0, stream>>>(Af, Wf, bias, out);
}
